// CNN_OTAM_43095701848257
// MI455X (gfx1250) — compile-verified, matrix operand order checked by reading, not run on hardware
//
#include <hip/hip_runtime.h>
#include <stdint.h>

// ---------------- problem geometry ----------------
#define QD     2048          // queries
#define SD     8             // frames per clip
#define DD     2048          // feature dim
#define SPD    100           // supports
#define NCLS   20
#define MROWS  (QD * SD)     // 16384
#define NROWS  (SPD * SD)    // 800
#define KDIM   DD            // 2048

#define LBDA_F     0.1f
#define INV_LBDA_F 10.0f
#define EPS_F      0.01f

// ---------------- GEMM tiling ----------------
#define MBLK   128
#define NBLK   64
#define KSTEP  32
#define KTILES (KDIM / KSTEP)   // 64
#define LDS_PAD 8               // shorts of padding per row

typedef __attribute__((ext_vector_type(16))) __bf16 bf16x16;
typedef __attribute__((ext_vector_type(8)))  float  floatx8;

struct alignas(16) U128 { unsigned int x, y, z, w; };
struct alignas(8)  US4  { unsigned short x, y, z, w; };

union FragU { U128 u[2]; bf16x16 v; };

__device__ __forceinline__ unsigned short f2bf(float f) {
    unsigned int u = __float_as_uint(f);
    u += 0x7fffu + ((u >> 16) & 1u);          // round-to-nearest-even
    return (unsigned short)(u >> 16);
}

// ---------------------------------------------------------------------------
// Kernel 1: fp32 -> bf16 conversion + row L2 norms (one block per row)
// ---------------------------------------------------------------------------
__global__ __launch_bounds__(256) void k_convert_norm(
    const float* __restrict__ src, unsigned short* __restrict__ dst,
    float* __restrict__ norms)
{
    const int row = blockIdx.x;
    const size_t base = (size_t)row * KDIM;
    const float4* s4 = (const float4*)(src + base);
    US4* d4 = (US4*)(dst + base);

    float ss = 0.0f;
    for (int i = threadIdx.x; i < KDIM / 4; i += 256) {
        float4 v = s4[i];
        ss = fmaf(v.x, v.x, ss);
        ss = fmaf(v.y, v.y, ss);
        ss = fmaf(v.z, v.z, ss);
        ss = fmaf(v.w, v.w, ss);
        US4 o;
        o.x = f2bf(v.x); o.y = f2bf(v.y); o.z = f2bf(v.z); o.w = f2bf(v.w);
        d4[i] = o;
    }
    __shared__ float red[256];
    red[threadIdx.x] = ss;
    __syncthreads();
    for (int s = 128; s > 0; s >>= 1) {
        if (threadIdx.x < s) red[threadIdx.x] += red[threadIdx.x + s];
        __syncthreads();
    }
    if (threadIdx.x == 0) norms[row] = sqrtf(red[0]);
}

// ---------------------------------------------------------------------------
// Kernel 2: bf16 WMMA GEMM, epilogue = 1 - dot/(|x||y|+eps), permuted store
// dists layout: [Q][Sp][ts][ss]  (ts,ss in 0..7)
// ---------------------------------------------------------------------------
__global__ __launch_bounds__(256) void k_gemm_wmma(
    const unsigned short* __restrict__ A,   // [16384][2048] bf16
    const unsigned short* __restrict__ B,   // [  800][2048] bf16
    const float* __restrict__ tnorm,
    const float* __restrict__ snorm,
    float* __restrict__ dists)
{
    __shared__ unsigned short ldsA[2][MBLK][KSTEP + LDS_PAD];
    __shared__ unsigned short ldsB[2][NBLK][KSTEP + LDS_PAD];

    const int tid  = threadIdx.x;
    const int lane = tid & 31;
    const int wave = tid >> 5;     // 0..7 (wave32)
    const int wm   = wave >> 1;    // 0..3 : 32 M-rows each
    const int wn   = wave & 1;     // 0..1 : 32 N-cols each
    const int m0   = blockIdx.x * MBLK;
    const int n0   = blockIdx.y * NBLK;

    const int khalf = lane >> 4;   // K-half select per ISA 16-bit A/B layout
    const int lrow  = lane & 15;

    // global->register staging assignments
    const int arow = tid >> 1, ahalf = tid & 1;   // A: 128 rows x 2 half-rows
    const int brow = tid >> 2, bq    = tid & 3;   // B:  64 rows x 4 quarters

    U128 ra0, ra1, rb;
    auto fetch = [&](int kt) {
        const size_t ka = (size_t)(m0 + arow) * KDIM + (size_t)kt * KSTEP + ahalf * 16;
        ra0 = *(const U128*)(A + ka);
        ra1 = *(const U128*)(A + ka + 8);
        const int gn = n0 + brow;
        if (gn < NROWS) {
            rb = *(const U128*)(B + (size_t)gn * KDIM + (size_t)kt * KSTEP + bq * 8);
        } else {
            rb.x = 0u; rb.y = 0u; rb.z = 0u; rb.w = 0u;
        }
    };
    auto stage = [&](int buf) {
        *(U128*)&ldsA[buf][arow][ahalf * 16]     = ra0;
        *(U128*)&ldsA[buf][arow][ahalf * 16 + 8] = ra1;
        *(U128*)&ldsB[buf][brow][bq * 8]         = rb;
    };

    floatx8 acc[2][2];
    #pragma unroll
    for (int i = 0; i < 2; ++i)
        #pragma unroll
        for (int j = 0; j < 2; ++j)
            #pragma unroll
            for (int e = 0; e < 8; ++e) acc[i][j][e] = 0.0f;

    fetch(0);
    stage(0);
    __syncthreads();

    for (int kt = 0; kt < KTILES; ++kt) {
        const int buf = kt & 1;
        if (kt + 1 < KTILES) fetch(kt + 1);   // overlap global latency with WMMA

        bf16x16 afr[2], bfr[2];
        #pragma unroll
        for (int mt = 0; mt < 2; ++mt) {
            const unsigned short* p = &ldsA[buf][wm * 32 + mt * 16 + lrow][0];
            FragU f;
            f.u[0] = *(const U128*)(p + khalf * 8);        // K 0..7  / 8..15
            f.u[1] = *(const U128*)(p + 16 + khalf * 8);   // K 16..23 / 24..31
            afr[mt] = f.v;
        }
        #pragma unroll
        for (int nt = 0; nt < 2; ++nt) {
            const unsigned short* p = &ldsB[buf][wn * 32 + nt * 16 + lrow][0];
            FragU f;
            f.u[0] = *(const U128*)(p + khalf * 8);
            f.u[1] = *(const U128*)(p + 16 + khalf * 8);
            bfr[nt] = f.v;
        }
        #pragma unroll
        for (int mt = 0; mt < 2; ++mt)
            #pragma unroll
            for (int nt = 0; nt < 2; ++nt)
                acc[mt][nt] = __builtin_amdgcn_wmma_f32_16x16x32_bf16(
                    false, afr[mt], false, bfr[nt], (short)0, acc[mt][nt],
                    false, false);

        if (kt + 1 < KTILES) stage((kt + 1) & 1);
        __syncthreads();
    }

    // epilogue: distance + permuted store [Q][Sp][8][8]
    #pragma unroll
    for (int nt = 0; nt < 2; ++nt) {
        const int nglob = n0 + wn * 32 + nt * 16 + lrow;
        if (nglob >= NROWS) continue;
        const float sn = snorm[nglob];
        const int sp = nglob >> 3, ssi = nglob & 7;
        #pragma unroll
        for (int mt = 0; mt < 2; ++mt) {
            #pragma unroll
            for (int e = 0; e < 8; ++e) {
                const int mglob = m0 + wm * 32 + mt * 16 + e + 8 * khalf;
                const float tn = tnorm[mglob];
                const float dist = 1.0f - acc[mt][nt][e] / (tn * sn + EPS_F);
                const int q = mglob >> 3, ts = mglob & 7;
                dists[(((size_t)q * SPD + sp) << 6) + ts * 8 + ssi] = dist;
            }
        }
    }
}

// ---------------------------------------------------------------------------
// Kernel 3: OTAM soft-DTW, one (q,sp) pair per lane, both directions
// ---------------------------------------------------------------------------
__device__ __forceinline__ float softmin2(float a, float b) {
    const float mn = fminf(a, b);
    const float df = fabsf(a - b);
    return mn - LBDA_F * __logf(1.0f + __expf(-df * INV_LBDA_F));
}
__device__ __forceinline__ float softmin3(float a, float b, float c) {
    const float mn = fminf(fminf(a, b), c);
    return mn - LBDA_F * __logf(__expf((mn - a) * INV_LBDA_F) +
                                __expf((mn - b) * INV_LBDA_F) +
                                __expf((mn - c) * INV_LBDA_F));
}

template <bool TR>
__device__ float otam_pass(const float* d) {
    float prev[10], cur[10];
    auto D = [&](int i, int j) { return TR ? d[j * 8 + i] : d[i * 8 + j]; };
    prev[0] = 0.0f;
    #pragma unroll
    for (int m = 1; m <= 8; ++m) prev[m] = prev[m - 1] + D(0, m - 1);
    prev[9] = prev[8];                       // + zero pad
    #pragma unroll
    for (int r = 1; r < 8; ++r) {
        cur[0] = 0.0f;
        cur[1] = D(r, 0) + softmin3(prev[0], prev[1], cur[0]);
        #pragma unroll
        for (int m = 2; m <= 8; ++m)
            cur[m] = D(r, m - 1) + softmin2(prev[m - 1], cur[m - 1]);
        cur[9] = softmin3(prev[8], prev[9], cur[8]);   // pad col: d=0
        #pragma unroll
        for (int m = 0; m < 10; ++m) prev[m] = cur[m];
    }
    return prev[9];
}

__global__ __launch_bounds__(256) void k_otam(
    const float* __restrict__ dists, float* __restrict__ cum)
{
    const int pair = blockIdx.x * blockDim.x + threadIdx.x;
    if (pair >= QD * SPD) return;
    float d[64];
    const float4* p = (const float4*)(dists + (size_t)pair * 64);
    #pragma unroll
    for (int i = 0; i < 16; ++i) {
        float4 v = p[i];
        d[4 * i] = v.x; d[4 * i + 1] = v.y; d[4 * i + 2] = v.z; d[4 * i + 3] = v.w;
    }
    cum[pair] = otam_pass<false>(d) + otam_pass<true>(d);
}

// ---------------------------------------------------------------------------
// Kernel 4: per-class mean + negate -> logits [1, Q, C]
// ---------------------------------------------------------------------------
__global__ __launch_bounds__(256) void k_class(
    const float* __restrict__ cum, const int* __restrict__ labels,
    float* __restrict__ out)
{
    const int t = blockIdx.x * blockDim.x + threadIdx.x;
    if (t >= QD * NCLS) return;
    const int q = t / NCLS, c = t % NCLS;
    float s = 0.0f; int cnt = 0;
    for (int sp = 0; sp < SPD; ++sp) {
        if (labels[sp] == c) { s += cum[q * SPD + sp]; ++cnt; }
    }
    out[t] = -s / (float)cnt;
}

// ---------------------------------------------------------------------------
extern "C" void kernel_launch(void* const* d_in, const int* in_sizes, int n_in,
                              void* d_out, int out_size, void* d_ws, size_t ws_size,
                              hipStream_t stream)
{
    (void)in_sizes; (void)n_in; (void)out_size; (void)ws_size;
    const float* tf     = (const float*)d_in[0];   // [2048,8,2048]
    const float* sf     = (const float*)d_in[1];   // [100,8,2048]
    const int*   labels = (const int*)d_in[2];     // [100]
    float* out = (float*)d_out;                    // [1,2048,20]

    // workspace layout (~124 MB, fits in 192 MB L2)
    char* ws = (char*)d_ws;
    size_t o = 0;
    unsigned short* tf_bf = (unsigned short*)(ws + o); o += (size_t)MROWS * KDIM * 2;
    unsigned short* sf_bf = (unsigned short*)(ws + o); o += (size_t)NROWS * KDIM * 2;
    float* tnorm = (float*)(ws + o);                   o += (size_t)MROWS * 4;
    float* snorm = (float*)(ws + o);                   o += 4096;   // 800*4 padded
    float* dists = (float*)(ws + o);                   o += (size_t)QD * SPD * 64 * 4;
    float* cum   = (float*)(ws + o);

    k_convert_norm<<<MROWS, 256, 0, stream>>>(tf, tf_bf, tnorm);
    k_convert_norm<<<NROWS, 256, 0, stream>>>(sf, sf_bf, snorm);

    dim3 grid(MROWS / MBLK, (NROWS + NBLK - 1) / NBLK);   // 128 x 13
    k_gemm_wmma<<<grid, 256, 0, stream>>>(tf_bf, sf_bf, tnorm, snorm, dists);

    k_otam<<<(QD * SPD + 255) / 256, 256, 0, stream>>>(dists, cum);
    k_class<<<(QD * NCLS + 255) / 256, 256, 0, stream>>>(cum, labels, out);
}
